// Block_44719199486321
// MI455X (gfx1250) — matrix operand order repaired and checked by reading, not compiled, not run on hardware
//
#include <hip/hip_runtime.h>
#include <stdint.h>

typedef __attribute__((ext_vector_type(16))) _Float16 v16h;
typedef __attribute__((ext_vector_type(8)))  float    v8f;
typedef __attribute__((ext_vector_type(4)))  unsigned int u32x4;
typedef __attribute__((ext_vector_type(8)))  int      i32x8;
typedef __attribute__((ext_vector_type(4)))  int      i32x4;
typedef _Float16 h16;

#define K_C    256
#define K_HID  512
#define K_H    64
#define K_B    32
#define K_NPIX (K_B * K_H * K_H)        /* 131072 */
#define K_WR   7
#define K_NWS  10                        /* 70/7 windows per side */
#define K_NWIN (K_B * K_NWS * K_NWS)     /* 3200 */
#define K_NTOKW 49
#define K_NTOK (K_NWIN * K_NTOKW)        /* 156800 */
#define K_HD   64

// ---- WMMA fragment index helpers (16-bit A 16x32, contiguous k = 16*(lane/16) + e, matching B) ----
__device__ __forceinline__ int a_frag_k(int lane, int e) {
    int kh = lane >> 4;
    return kh * 16 + e;
}

// ---- Tensor Data Mover: DMA a 2D f16 tile (tile0 x tile1) of a row-major
// tensor (row stride = stride0 elements) from global into LDS (packed rows).
// D# layout per CDNA5 ISA.
__device__ __forceinline__ void tdm_load_2d_f16(uint32_t lds_addr,
                                                const void* gaddr,
                                                uint32_t tdim0, uint32_t tdim1,
                                                uint32_t tile0, uint32_t tile1,
                                                uint64_t stride0) {
    uint64_t ga = (uint64_t)(uintptr_t)gaddr;
    u32x4 g0;
    g0[0] = 1u;                                              // count=1, user mode
    g0[1] = lds_addr;                                        // lds_addr [63:32]
    g0[2] = (uint32_t)ga;                                    // global_addr lo
    g0[3] = (uint32_t)((ga >> 32) & 0x01FFFFFFu) | (2u << 30); // addr hi | type=2
    i32x8 g1;
    g1[0] = (int)(1u << 16);                                 // data_size=1 (2B)
    g1[1] = (int)((tdim0 & 0xFFFFu) << 16);                  // tensor_dim0 lo16
    g1[2] = (int)(((tdim0 >> 16) & 0xFFFFu) | ((tdim1 & 0xFFFFu) << 16));
    g1[3] = (int)(((tdim1 >> 16) & 0xFFFFu) | (tile0 << 16)); // tile_dim0
    g1[4] = (int)(tile1 & 0xFFFFu);                          // tile_dim1 (dim2=0)
    g1[5] = (int)(uint32_t)(stride0 & 0xFFFFFFFFu);          // dim0 stride lo
    g1[6] = (int)(uint32_t)((stride0 >> 32) & 0xFFFFu);      // dim0 stride hi
    g1[7] = 0;                                               // dim1 stride = 0
    i32x4 z4 = {0, 0, 0, 0};
#if defined(__clang_major__) && (__clang_major__ >= 23)
    i32x8 z8 = {0, 0, 0, 0, 0, 0, 0, 0};
    __builtin_amdgcn_tensor_load_to_lds(g0, g1, z4, z4, z8, 0);
#else
    __builtin_amdgcn_tensor_load_to_lds(g0, g1, z4, z4, 0);
#endif
}

// ---------------------------------------------------------------------------
// NCHW -> NHWC fp32
__global__ __launch_bounds__(256) void k_nchw2nhwc(const float* __restrict__ in,
                                                   float* __restrict__ out) {
    int i = blockIdx.x * 256 + threadIdx.x;          // over NPIX*C
    int c = i & 255;
    int p = i >> 8;
    int w = p & 63, h = (p >> 6) & 63, b = p >> 12;
    out[i] = in[(((b * K_C) + c) * K_H + h) * K_H + w];
}

// fp32 -> f16 copy
__global__ __launch_bounds__(256) void k_tof16(const float* __restrict__ in,
                                               h16* __restrict__ out, int n) {
    int i = blockIdx.x * 256 + threadIdx.x;
    if (i < n) out[i] = (h16)in[i];
}

// depthwise 3x3 + BN + residual, NHWC
__global__ __launch_bounds__(256) void k_dw3x3(const float* __restrict__ in,
                                               const float* __restrict__ wgt,
                                               const float* __restrict__ sc,
                                               const float* __restrict__ bi,
                                               float* __restrict__ out) {
    int i = blockIdx.x * 256 + threadIdx.x;
    int c = i & 255;
    int p = i >> 8;
    int x = p & 63, y = (p >> 6) & 63, b = p >> 12;
    float acc = 0.f;
    for (int dy = -1; dy <= 1; ++dy) {
        int yy = y + dy;
        if (yy < 0 || yy >= K_H) continue;
        for (int dx = -1; dx <= 1; ++dx) {
            int xx = x + dx;
            if (xx < 0 || xx >= K_H) continue;
            acc += in[(((b * K_H) + yy) * K_H + xx) * K_C + c] *
                   wgt[c * 9 + (dy + 1) * 3 + (dx + 1)];
        }
    }
    out[i] = in[i] + sc[c] * acc + bi[c];
}

// gather padded-window token matrix (zero pad outside image), NHWC f32 -> f16
__global__ __launch_bounds__(256) void k_wingather(const float* __restrict__ in,
                                                   h16* __restrict__ out) {
    int i = blockIdx.x * 256 + threadIdx.x;          // over NTOK*C
    int c = i & 255;
    int t = i >> 8;
    int n   = t % K_NTOKW;
    int win = t / K_NTOKW;
    int b  = win / (K_NWS * K_NWS);
    int wh = (win / K_NWS) % K_NWS;
    int ww = win % K_NWS;
    int ph = wh * K_WR + n / K_WR;
    int pw = ww * K_WR + n % K_WR;
    float v = 0.f;
    if (ph < K_H && pw < K_H)
        v = in[(((b * K_H) + ph) * K_H + pw) * K_C + c];
    out[i] = (h16)v;
}

// focused linear attention nonlinearity, one warp per token (wave32)
__global__ __launch_bounds__(256) void k_focus_q(h16* __restrict__ Q,
                                                 const float* __restrict__ scale_p) {
    int tok  = (blockIdx.x * 256 + threadIdx.x) >> 5;
    int lane = threadIdx.x & 31;
    h16* row = Q + (size_t)tok * K_C;
    float t[8];
    float ss = 0.f;
    for (int i = 0; i < 8; ++i) {
        int c = lane + 32 * i;
        float scl = log1pf(expf(scale_p[c]));        // softplus
        float v = (float)row[c];
        v = (fmaxf(v, 0.f) + 1e-6f) / scl;
        t[i] = v;
        ss += v * v;
    }
    for (int m = 16; m; m >>= 1) ss += __shfl_xor(ss, m);
    float n0 = sqrtf(ss);
    float sp = 0.f;
    for (int i = 0; i < 8; ++i) { float p = t[i] * t[i] * t[i]; t[i] = p; sp += p * p; }
    for (int m = 16; m; m >>= 1) sp += __shfl_xor(sp, m);
    float g = n0 / sqrtf(sp);
    for (int i = 0; i < 8; ++i) row[lane + 32 * i] = (h16)(t[i] * g);
}

__global__ __launch_bounds__(256) void k_focus_k(const h16* __restrict__ KV,
                                                 const float* __restrict__ pos,
                                                 const float* __restrict__ scale_p,
                                                 h16* __restrict__ Kf) {
    int tok  = (blockIdx.x * 256 + threadIdx.x) >> 5;
    int lane = threadIdx.x & 31;
    int n = tok % K_NTOKW;
    float t[8];
    float ss = 0.f;
    for (int i = 0; i < 8; ++i) {
        int c = lane + 32 * i;
        float scl = log1pf(expf(scale_p[c]));
        float v = (float)KV[(size_t)tok * 512 + c] + pos[n * K_C + c];
        v = (fmaxf(v, 0.f) + 1e-6f) / scl;
        t[i] = v;
        ss += v * v;
    }
    for (int m = 16; m; m >>= 1) ss += __shfl_xor(ss, m);
    float n0 = sqrtf(ss);
    float sp = 0.f;
    for (int i = 0; i < 8; ++i) { float p = t[i] * t[i] * t[i]; t[i] = p; sp += p * p; }
    for (int m = 16; m; m >>= 1) sp += __shfl_xor(sp, m);
    float g = n0 / sqrtf(sp);
    for (int i = 0; i < 8; ++i)
        Kf[(size_t)tok * K_C + lane + 32 * i] = (h16)(t[i] * g);
}

// ---------------------------------------------------------------------------
// Tiled WMMA GEMM with TDM double-buffered tile loads.
// C[M,N] = epilogue(A[M,K] @ B[N,K]^T)
// mode 0: hout = f16(relu?(val*s+b))
// mode 1: fout = res + val*s+b                     (FFN residual, f32 NHWC)
// mode 2: fout[pixel] += val + b  (token scatter)  (attention proj residual)
// mode 3: d_out NCHW = res + val*s+b               (final write)
__global__ __launch_bounds__(256) void k_gemm(const h16* __restrict__ A,
                                              const h16* __restrict__ B,
                                              int M, int N, int K,
                                              const float* __restrict__ scale,
                                              const float* __restrict__ bias,
                                              int relu, int mode,
                                              const float* __restrict__ res,
                                              float* __restrict__ fout,
                                              h16* __restrict__ hout) {
    __shared__ __align__(16) h16 As[2][128 * 32];   // TDM-packed, row stride 32
    __shared__ __align__(16) h16 Bs[2][64 * 32];

    const int m0 = blockIdx.x * 128;
    const int n0 = blockIdx.y * 64;
    const int tid = threadIdx.x, lane = tid & 31, wave = tid >> 5;
    const int wm0 = (wave & 3) * 32;
    const int wn0 = (wave >> 2) * 32;

    v8f acc[2][2] = {};

    const uint32_t ldsA0 = (uint32_t)(uintptr_t)&As[0][0];
    const uint32_t ldsA1 = (uint32_t)(uintptr_t)&As[1][0];
    const uint32_t ldsB0 = (uint32_t)(uintptr_t)&Bs[0][0];
    const uint32_t ldsB1 = (uint32_t)(uintptr_t)&Bs[1][0];

    // preload tile 0 via Tensor Data Mover (wave 0 programs the D#)
    if (wave == 0) {
        tdm_load_2d_f16(ldsA0, A + (size_t)m0 * K, (uint32_t)K, (uint32_t)M,
                        32u, 128u, (uint64_t)K);
        tdm_load_2d_f16(ldsB0, B + (size_t)n0 * K, (uint32_t)K, (uint32_t)N,
                        32u, 64u, (uint64_t)K);
    }

    int buf = 0;
    for (int k0 = 0; k0 < K; k0 += 32) {
        if (wave == 0) __builtin_amdgcn_s_wait_tensorcnt(0);
        __syncthreads();   // tile `buf` visible to all waves; prev compute done
        if (wave == 0 && (k0 + 32) < K) {
            tdm_load_2d_f16(buf ? ldsA0 : ldsA1,
                            A + (size_t)m0 * K + k0 + 32, (uint32_t)K,
                            (uint32_t)M, 32u, 128u, (uint64_t)K);
            tdm_load_2d_f16(buf ? ldsB0 : ldsB1,
                            B + (size_t)n0 * K + k0 + 32, (uint32_t)K,
                            (uint32_t)N, 32u, 64u, (uint64_t)K);
        }

        v16h af[2], bf[2];
        for (int mi = 0; mi < 2; ++mi) {
            int row = wm0 + mi * 16 + (lane & 15);
            for (int e = 0; e < 16; ++e)
                af[mi][e] = As[buf][row * 32 + a_frag_k(lane, e)];
        }
        for (int ni = 0; ni < 2; ++ni) {
            int col = wn0 + ni * 16 + (lane & 15);
            int kb  = (lane >> 4) * 16;
            for (int e = 0; e < 16; ++e)
                bf[ni][e] = Bs[buf][col * 32 + kb + e];
        }
        for (int mi = 0; mi < 2; ++mi)
            for (int ni = 0; ni < 2; ++ni)
                acc[mi][ni] = __builtin_amdgcn_wmma_f32_16x16x32_f16(
                    false, af[mi], false, bf[ni], (short)0, acc[mi][ni], false, false);
        buf ^= 1;
    }

    for (int mi = 0; mi < 2; ++mi)
        for (int ni = 0; ni < 2; ++ni)
            for (int r = 0; r < 8; ++r) {
                int mg = m0 + wm0 + mi * 16 + r + 8 * (lane >> 4);
                int ng = n0 + wn0 + ni * 16 + (lane & 15);
                float val = acc[mi][ni][r];
                float s = scale ? scale[ng] : 1.f;
                float b = bias ? bias[ng] : 0.f;
                if (mode == 0) {
                    val = val * s + b;
                    if (relu) val = fmaxf(val, 0.f);
                    hout[(size_t)mg * N + ng] = (h16)val;
                } else if (mode == 1) {
                    fout[(size_t)mg * N + ng] = res[(size_t)mg * N + ng] + val * s + b;
                } else if (mode == 2) {
                    int n_in = mg % K_NTOKW;
                    int win  = mg / K_NTOKW;
                    int bb = win / (K_NWS * K_NWS);
                    int wh = (win / K_NWS) % K_NWS;
                    int ww = win % K_NWS;
                    int ph = wh * K_WR + n_in / K_WR;
                    int pw = ww * K_WR + n_in % K_WR;
                    if (ph < K_H && pw < K_H) {
                        size_t idx = (size_t)(((bb * K_H) + ph) * K_H + pw) * K_C + ng;
                        fout[idx] += val + b;
                    }
                } else { // mode 3: NCHW final
                    int pb = mg >> 12, ph = (mg >> 6) & 63, pw = mg & 63;
                    fout[(size_t)(((pb * K_C) + ng) * K_H + ph) * K_H + pw] =
                        res[(size_t)mg * N + ng] + val * s + b;
                }
            }
}

// ---------------------------------------------------------------------------
// Per-window linear attention: qk = q k^T, y = z * (qk v) + dwconv5x5(v)
// 8 waves: 2 waves per head, each wave owns 32 of 64 (padded) rows.
__global__ __launch_bounds__(256) void k_attn(const h16* __restrict__ Q,
                                              const h16* __restrict__ Kf,
                                              const h16* __restrict__ KV,
                                              const float* __restrict__ dwcw,
                                              const float* __restrict__ dwcb,
                                              h16* __restrict__ Y) {
    const int win  = blockIdx.x;
    const int tid  = threadIdx.x, lane = tid & 31, wave = tid >> 5;
    const int head = wave >> 1;
    const int wrow = (wave & 1) * 32;
    const int tok0 = win * K_NTOKW;

    __shared__ float ksumS[256];
    __shared__ float zS[256];
    __shared__ __align__(16) h16 qkS[4][64][72];

    // ksum over all 49 (incl. pad) focused-k tokens
    {
        float s = 0.f;
        for (int j = 0; j < K_NTOKW; ++j) s += (float)Kf[(size_t)(tok0 + j) * K_C + tid];
        ksumS[tid] = s;
    }
    __syncthreads();
    // z[head][i] = 1/(q_i . ksum_head + eps)
    {
        int i = tid & 63, hh = tid >> 6;
        float z = 0.f;
        if (i < K_NTOKW) {
            float d = 0.f;
            for (int c = 0; c < K_HD; ++c)
                d += (float)Q[(size_t)(tok0 + i) * K_C + hh * K_HD + c] * ksumS[hh * K_HD + c];
            z = 1.f / (d + 1e-6f);
        }
        zS[tid] = z;
    }

    const v8f zerov = {};
    v8f acc[2][4] = {};
    // GEMM1: qk (64x64 padded), K-dim = hd = 64 -> 2 steps of 32
    for (int s = 0; s < 2; ++s) {
        v16h af[2], bf[4];
        for (int mi = 0; mi < 2; ++mi) {
            int i = wrow + mi * 16 + (lane & 15);
            for (int e = 0; e < 16; ++e) {
                int c = head * K_HD + s * 32 + a_frag_k(lane, e);
                af[mi][e] = (i < K_NTOKW) ? Q[(size_t)(tok0 + i) * K_C + c] : (h16)0.f;
            }
        }
        for (int ni = 0; ni < 4; ++ni) {
            int j  = ni * 16 + (lane & 15);
            int kb = (lane >> 4) * 16;
            for (int e = 0; e < 16; ++e) {
                int c = head * K_HD + s * 32 + kb + e;
                bf[ni][e] = (j < K_NTOKW) ? Kf[(size_t)(tok0 + j) * K_C + c] : (h16)0.f;
            }
        }
        for (int mi = 0; mi < 2; ++mi)
            for (int ni = 0; ni < 4; ++ni)
                acc[mi][ni] = __builtin_amdgcn_wmma_f32_16x16x32_f16(
                    false, af[mi], false, bf[ni], (short)0, acc[mi][ni], false, false);
    }
    for (int mi = 0; mi < 2; ++mi)
        for (int ni = 0; ni < 4; ++ni)
            for (int r = 0; r < 8; ++r) {
                int m = wrow + mi * 16 + r + 8 * (lane >> 4);
                int j = ni * 16 + (lane & 15);
                qkS[head][m][j] = (h16)acc[mi][ni][r];
            }
    __syncthreads();

    // GEMM2: y = qk @ v  (64x64, K = 64 padded tokens)
    v8f acc2[2][4];
    for (int mi = 0; mi < 2; ++mi)
        for (int ni = 0; ni < 4; ++ni) acc2[mi][ni] = zerov;
    for (int s = 0; s < 2; ++s) {
        v16h af[2], bf[4];
        for (int mi = 0; mi < 2; ++mi) {
            int i = wrow + mi * 16 + (lane & 15);
            for (int e = 0; e < 16; ++e)
                af[mi][e] = qkS[head][i][s * 32 + a_frag_k(lane, e)];
        }
        for (int ni = 0; ni < 4; ++ni) {
            int d  = ni * 16 + (lane & 15);
            int kb = (lane >> 4) * 16;
            for (int e = 0; e < 16; ++e) {
                int j = s * 32 + kb + e;
                bf[ni][e] = (j < K_NTOKW)
                                ? KV[(size_t)(tok0 + j) * 512 + 256 + head * K_HD + d]
                                : (h16)0.f;
            }
        }
        for (int mi = 0; mi < 2; ++mi)
            for (int ni = 0; ni < 4; ++ni)
                acc2[mi][ni] = __builtin_amdgcn_wmma_f32_16x16x32_f16(
                    false, af[mi], false, bf[ni], (short)0, acc2[mi][ni], false, false);
    }

    // epilogue: scale by z, add per-window 5x5 depthwise conv of v, store f16
    for (int mi = 0; mi < 2; ++mi)
        for (int ni = 0; ni < 4; ++ni)
            for (int r = 0; r < 8; ++r) {
                int i = wrow + mi * 16 + r + 8 * (lane >> 4);
                if (i >= K_NTOKW) continue;
                int d = ni * 16 + (lane & 15);
                float val = acc2[mi][ni][r] * zS[head * K_HD + i];
                int ry = i / K_WR, rx = i % K_WR;
                float conv = 0.f;
                for (int ky = 0; ky < 5; ++ky) {
                    int yy = ry + ky - 2;
                    if (yy < 0 || yy >= K_WR) continue;
                    for (int kx = 0; kx < 5; ++kx) {
                        int xx = rx + kx - 2;
                        if (xx < 0 || xx >= K_WR) continue;
                        int j2 = yy * K_WR + xx;
                        conv += (float)KV[(size_t)(tok0 + j2) * 512 + 256 + head * K_HD + d] *
                                dwcw[d * 25 + ky * 5 + kx];
                    }
                }
                val += conv + dwcb[d];
                Y[(size_t)(tok0 + i) * K_C + head * K_HD + d] = (h16)val;
            }
}

// ---------------------------------------------------------------------------
extern "C" void kernel_launch(void* const* d_in, const int* in_sizes, int n_in,
                              void* d_out, int out_size, void* d_ws, size_t ws_size,
                              hipStream_t stream) {
    (void)in_sizes; (void)n_in; (void)out_size; (void)ws_size;

    const float* x      = (const float*)d_in[0];
    const float* dw0_w  = (const float*)d_in[1];
    const float* dw0_s  = (const float*)d_in[2];
    const float* dw0_b  = (const float*)d_in[3];
    const float* f0_w1  = (const float*)d_in[4];
    const float* f0_s1  = (const float*)d_in[5];
    const float* f0_b1  = (const float*)d_in[6];
    const float* f0_w2  = (const float*)d_in[7];
    const float* f0_s2  = (const float*)d_in[8];
    const float* f0_b2  = (const float*)d_in[9];
    const float* wq     = (const float*)d_in[10];
    const float* wkv    = (const float*)d_in[11];
    const float* pos    = (const float*)d_in[12];
    const float* scp    = (const float*)d_in[13];
    const float* dwc_w  = (const float*)d_in[14];
    const float* dwc_b  = (const float*)d_in[15];
    const float* proj_w = (const float*)d_in[16];
    const float* proj_b = (const float*)d_in[17];
    const float* dw1_w  = (const float*)d_in[18];
    const float* dw1_s  = (const float*)d_in[19];
    const float* dw1_b  = (const float*)d_in[20];
    const float* f1_w1  = (const float*)d_in[21];
    const float* f1_s1  = (const float*)d_in[22];
    const float* f1_b1  = (const float*)d_in[23];
    const float* f1_w2  = (const float*)d_in[24];
    const float* f1_s2  = (const float*)d_in[25];
    const float* f1_b2  = (const float*)d_in[26];

    char* ws = (char*)d_ws;
    const size_t szMaster = (size_t)K_NPIX * K_C * 4;        // 134,217,728
    const size_t szBigA   = (size_t)K_NTOK * 512 * 2;        // 160,563,200
    const size_t szBigB   = (size_t)K_NPIX * K_HID * 2;      // 134,217,728
    float* Xa  = (float*)ws;
    float* Xb  = (float*)(ws + szMaster);
    h16* bigA  = (h16*)(ws + 2 * szMaster);
    h16* bigB  = (h16*)(ws + 2 * szMaster + szBigA);
    h16* bigC  = (h16*)(ws + 2 * szMaster + szBigA + szBigB);
    h16* wb    = (h16*)(ws + 2 * szMaster + 2 * szBigA + szBigB);

    // stage aliases
    h16* X16  = bigA;                     // FFN A-operand (NPIX x 256)
    h16* KV16 = bigA;                     // KV raw (NTOK x 512)
    h16* H16  = bigB;                     // FFN hidden (NPIX x 512)
    h16* Q16  = bigB;                     // Q (NTOK x 256), focused in place
    h16* Aw16 = bigC;                     // windowed tokens (NTOK x 256)
    h16* Y16  = bigC;                     // attn out pre-proj (reuses Aw16)
    h16* Kf16 = bigC + (size_t)K_NTOK * 256;

    h16* w1a   = wb;
    h16* w2a   = w1a + 512 * 256;
    h16* wq16  = w2a + 512 * 256;
    h16* wkv16 = wq16 + 256 * 256;
    h16* pj16  = wkv16 + 512 * 256;
    h16* w1b   = pj16 + 256 * 256;
    h16* w2b   = w1b + 512 * 256;

    const int nElem = K_NPIX * K_C;       // 33,554,432
    dim3 blk(256);

    k_nchw2nhwc<<<nElem / 256, blk, 0, stream>>>(x, Xa);

    k_tof16<<<512, blk, 0, stream>>>(f0_w1, w1a, 512 * 256);
    k_tof16<<<512, blk, 0, stream>>>(f0_w2, w2a, 512 * 256);
    k_tof16<<<256, blk, 0, stream>>>(wq, wq16, 256 * 256);
    k_tof16<<<512, blk, 0, stream>>>(wkv, wkv16, 512 * 256);
    k_tof16<<<256, blk, 0, stream>>>(proj_w, pj16, 256 * 256);
    k_tof16<<<512, blk, 0, stream>>>(f1_w1, w1b, 512 * 256);
    k_tof16<<<512, blk, 0, stream>>>(f1_w2, w2b, 512 * 256);

    // x += BN(dwconv3x3(x))
    k_dw3x3<<<nElem / 256, blk, 0, stream>>>(Xa, dw0_w, dw0_s, dw0_b, Xb);

    // x += FFN0(x)
    k_tof16<<<nElem / 256, blk, 0, stream>>>(Xb, X16, nElem);
    k_gemm<<<dim3(K_NPIX / 128, K_HID / 64), blk, 0, stream>>>(
        X16, w1a, K_NPIX, K_HID, K_C, f0_s1, f0_b1, 1, 0, nullptr, nullptr, H16);
    k_gemm<<<dim3(K_NPIX / 128, K_C / 64), blk, 0, stream>>>(
        H16, w2a, K_NPIX, K_C, K_HID, f0_s2, f0_b2, 0, 1, Xb, Xa, nullptr);

    // windowed focused linear attention
    k_wingather<<<(K_NTOK * K_C) / 256, blk, 0, stream>>>(Xa, Aw16);
    k_gemm<<<dim3(K_NTOK / 128, K_C / 64), blk, 0, stream>>>(
        Aw16, wq16, K_NTOK, K_C, K_C, nullptr, nullptr, 0, 0, nullptr, nullptr, Q16);
    k_gemm<<<dim3(K_NTOK / 128, K_HID / 64), blk, 0, stream>>>(
        Aw16, wkv16, K_NTOK, K_HID, K_C, nullptr, nullptr, 0, 0, nullptr, nullptr, KV16);
    k_focus_q<<<K_NTOK / 8, blk, 0, stream>>>(Q16, scp);
    k_focus_k<<<K_NTOK / 8, blk, 0, stream>>>(KV16, pos, scp, Kf16);
    k_attn<<<K_NWIN, blk, 0, stream>>>(Q16, Kf16, KV16, dwc_w, dwc_b, Y16);
    k_gemm<<<dim3(K_NTOK / 128, K_C / 64), blk, 0, stream>>>(
        Y16, pj16, K_NTOK, K_C, K_C, nullptr, proj_b, 0, 2, nullptr, Xa, nullptr);

    // x += BN(dwconv3x3(x))
    k_dw3x3<<<nElem / 256, blk, 0, stream>>>(Xa, dw1_w, dw1_s, dw1_b, Xb);

    // x += FFN1(x), write final NCHW into d_out
    k_tof16<<<nElem / 256, blk, 0, stream>>>(Xb, X16, nElem);
    k_gemm<<<dim3(K_NPIX / 128, K_HID / 64), blk, 0, stream>>>(
        X16, w1b, K_NPIX, K_HID, K_C, f1_s1, f1_b1, 1, 0, nullptr, nullptr, H16);
    k_gemm<<<dim3(K_NPIX / 128, K_C / 64), blk, 0, stream>>>(
        H16, w2b, K_NPIX, K_C, K_HID, f1_s2, f1_b2, 0, 3, Xb, (float*)d_out, nullptr);
}
